// AttnDecoderLSTM_78872779424165
// MI455X (gfx1250) — matrix operand order repaired and checked by reading, not compiled, not run on hardware
//
#include <hip/hip_runtime.h>
#include <stddef.h>

// ---------------- problem constants ----------------
#define H_  1024
#define V_  32000
#define B_  64
#define S_  64
#define T_  13
#define SOS_ 2

// ---------------- WMMA types ----------------
typedef __attribute__((ext_vector_type(16))) __bf16 v16bf;
typedef __attribute__((ext_vector_type(8)))  float  v8f;

union V16 {
    v16bf bf;
    uint4 u[2];
};

// round-to-nearest-even fp32 -> bf16 (as raw u16)
__device__ __forceinline__ unsigned short f2bf(float f) {
    unsigned int x = __float_as_uint(f);
    unsigned int r = x + 0x7FFFu + ((x >> 16) & 1u);
    return (unsigned short)(r >> 16);
}

__device__ __forceinline__ float sigmoidf_(float x) {
    return 1.f / (1.f + __expf(-x));
}

// ---- A tile loader: 16x32 bf16, A is [M,K] row-major (u16 storage) ----
// lane l (0..15): M=l,  elems 0..7 = K(k0+0..7),  8..15 = K(k0+16..23)
// lane l (16..31): M=l-16, elems 0..7 = K(k0+8..15), 8..15 = K(k0+24..31)
__device__ __forceinline__ v16bf load_a_tile(const unsigned short* __restrict__ A,
                                             int lda, int m0, int k0, int lane) {
    int m  = m0 + (lane & 15);
    int kh = (lane >> 4) << 3;          // 0 or 8
    const unsigned short* p = A + (size_t)m * lda + k0;
    V16 v;
    v.u[0] = *(const uint4*)(p + kh);        // K = k0+kh   .. +7
    v.u[1] = *(const uint4*)(p + 16 + kh);   // K = k0+16+kh.. +7
    return v.bf;
}

// ---- B tile loader: 32x16 bf16 from weight W [N,K] row-major (B = W^T) ----
// lane l: N = l&15, elems 0..7 = K(k0 + (l>>4)*8 + 0..7), 8..15 = K(k0 + 16 + (l>>4)*8 + 0..7)
__device__ __forceinline__ v16bf load_b_tile(const unsigned short* __restrict__ W,
                                             int ldb, int n0, int k0, int lane) {
    int n  = n0 + (lane & 15);
    int kq = (lane >> 4) << 3;          // 0 or 8
    const unsigned short* p = W + (size_t)n * ldb + k0 + kq;
    V16 v;
    v.u[0] = *(const uint4*)p;
    v.u[1] = *(const uint4*)(p + 16);
    return v.bf;
}

// ---------------- bf16 GEMM: C[M,N] = A[M,K] @ W[N,K]^T (+bias) ----------------
// One wave computes a 64x16 output strip (4 M-tiles x 1 N-tile):
//  - streaming operand B (the weight) is loaded ONCE per K-step and reused by
//    4 independent v_wmma_f32_16x16x32_bf16 (hides XDL latency, 4x less B traffic)
//  - A is tiny (M=64 rows for the per-step GEMMs) and stays L0/L2-resident
// Requires M % 64 == 0, N % 16 == 0, K % 32 == 0 (all shapes here qualify).
__global__ void k_gemm_bf16(const unsigned short* __restrict__ A, int lda,
                            const unsigned short* __restrict__ Bw, int ldb,
                            float* __restrict__ C, int ldc,
                            int M, int N, int K,
                            const float* __restrict__ bias0,
                            const float* __restrict__ bias1) {
    int wave = (int)((blockIdx.x * blockDim.x + threadIdx.x) >> 5);
    int lane = (int)(threadIdx.x & 31);
    int ntiles  = N >> 4;
    int mstrips = M >> 6;
    int total   = mstrips * ntiles;
    if (wave >= total) return;                 // wave-uniform exit (EXEC stays all-1s)
    int ms = wave / ntiles;                    // consecutive waves walk consecutive N
    int nt = wave - ms * ntiles;
    int m0 = ms << 6, n0 = nt << 4;

    v8f acc0 = {}, acc1 = {}, acc2 = {}, acc3 = {};
    for (int k0 = 0; k0 < K; k0 += 32) {
        v16bf b  = load_b_tile(Bw, ldb, n0, k0, lane);
        v16bf a0 = load_a_tile(A, lda, m0,      k0, lane);
        v16bf a1 = load_a_tile(A, lda, m0 + 16, k0, lane);
        v16bf a2 = load_a_tile(A, lda, m0 + 32, k0, lane);
        v16bf a3 = load_a_tile(A, lda, m0 + 48, k0, lane);
        if (k0 + 32 < K) {  // prefetch next K-chunk of the streaming operand
            __builtin_prefetch(Bw + (size_t)(n0 + (lane & 15)) * ldb + k0 + 32, 0, 3);
        }
        acc0 = __builtin_amdgcn_wmma_f32_16x16x32_bf16(false, a0, false, b, (short)0, acc0, false, false);
        acc1 = __builtin_amdgcn_wmma_f32_16x16x32_bf16(false, a1, false, b, (short)0, acc1, false, false);
        acc2 = __builtin_amdgcn_wmma_f32_16x16x32_bf16(false, a2, false, b, (short)0, acc2, false, false);
        acc3 = __builtin_amdgcn_wmma_f32_16x16x32_bf16(false, a3, false, b, (short)0, acc3, false, false);
    }

    // C/D layout: lane l holds N = l&15; VGPR i holds M = tile_m0 + (l>>4)*8 + i
    int n  = n0 + (lane & 15);
    int mb = m0 + ((lane >> 4) << 3);
    float bv = 0.f;
    if (bias0) bv += bias0[n];
    if (bias1) bv += bias1[n];
#pragma unroll
    for (int i = 0; i < 8; ++i) {
        C[(size_t)(mb      + i) * ldc + n] = acc0[i] + bv;
        C[(size_t)(mb + 16 + i) * ldc + n] = acc1[i] + bv;
        C[(size_t)(mb + 32 + i) * ldc + n] = acc2[i] + bv;
        C[(size_t)(mb + 48 + i) * ldc + n] = acc3[i] + bv;
    }
}

// ---------------- conversion / setup kernels ----------------
__global__ void k_f32_to_bf16(const float* __restrict__ src,
                              unsigned short* __restrict__ dst, size_t n) {
    size_t i = (size_t)blockIdx.x * blockDim.x + threadIdx.x;
    size_t stride = (size_t)gridDim.x * blockDim.x;
    for (; i < n; i += stride) dst[i] = f2bf(src[i]);
}

// Wcat[4H, 3H] = [ W_ih (4H,2H) | W_hh (4H,H) ] in bf16
__global__ void k_build_wcat(const float* __restrict__ Wih,
                             const float* __restrict__ Whh,
                             unsigned short* __restrict__ Wcat) {
    size_t n = (size_t)(4 * H_) * (3 * H_);
    size_t i = (size_t)blockIdx.x * blockDim.x + threadIdx.x;
    size_t stride = (size_t)gridDim.x * blockDim.x;
    for (; i < n; i += stride) {
        int row = (int)(i / (3 * H_));
        int col = (int)(i - (size_t)row * (3 * H_));
        float v = (col < 2 * H_) ? Wih[(size_t)row * (2 * H_) + col]
                                 : Whh[(size_t)row * H_ + (col - 2 * H_)];
        Wcat[i] = f2bf(v);
    }
}

__global__ void k_init_state(const float* __restrict__ eh, const float* __restrict__ ec,
                             float* __restrict__ hf, float* __restrict__ cf,
                             unsigned short* __restrict__ hb) {
    int i = blockIdx.x * blockDim.x + threadIdx.x;
    if (i >= B_ * H_) return;
    float h = eh[i];
    hf[i] = h;
    cf[i] = ec[i];
    hb[i] = f2bf(h);
}

// ---------------- attention: scores(tanh) + softmax + ctx, one block per batch ----------------
__global__ void k_attn(const float* __restrict__ qf, const float* __restrict__ keys,
                       const float* __restrict__ Va, const float* __restrict__ bv,
                       const float* __restrict__ enc, float* __restrict__ ctx,
                       float* __restrict__ attn_out, int t) {
    __shared__ float sQ[H_];
    __shared__ float sV[H_];
    __shared__ float sSc[S_];
    __shared__ float sW[S_];
    __shared__ float sInv;

    int b = blockIdx.x;
    int tid = threadIdx.x;                 // 256 threads = 8 waves
    for (int i = tid; i < H_; i += 256) {
        sQ[i] = qf[(size_t)b * H_ + i];
        sV[i] = Va[i];
    }
    __syncthreads();

    int wv = tid >> 5, ln = tid & 31;
    for (int j = 0; j < 8; ++j) {          // each wave: 8 source positions
        int s = wv * 8 + j;
        const float* kp = keys + ((size_t)b * S_ + s) * H_;
        float acc = 0.f;
        for (int h = ln; h < H_; h += 32)
            acc += tanhf(sQ[h] + kp[h]) * sV[h];
        for (int off = 16; off; off >>= 1)
            acc += __shfl_down(acc, off, 32);
        if (ln == 0) sSc[s] = acc + bv[0];
    }
    __syncthreads();

    if (tid == 0) {                        // softmax over S=64 (tiny)
        float m = -3.4e38f;
        for (int s = 0; s < S_; ++s) m = fmaxf(m, sSc[s]);
        float sum = 0.f;
        for (int s = 0; s < S_; ++s) { float e = __expf(sSc[s] - m); sW[s] = e; sum += e; }
        sInv = 1.f / sum;
    }
    __syncthreads();
    float inv = sInv;
    for (int s = tid; s < S_; s += 256) {
        float w = sW[s] * inv;
        sW[s] = w;
        attn_out[((size_t)b * T_ + t) * S_ + s] = w;
    }
    __syncthreads();

    for (int h = tid; h < H_; h += 256) {  // ctx = w @ enc[b]
        float acc = 0.f;
        const float* eb = enc + (size_t)b * S_ * H_ + h;
        for (int s = 0; s < S_; ++s) acc += sW[s] * eb[(size_t)s * H_];
        ctx[(size_t)b * H_ + h] = acc;
    }
}

// ---------------- pack xcat = [emb[tok], ctx, h] in bf16, one block per batch ----------------
__global__ void k_pack(const int* __restrict__ target, const float* __restrict__ emb,
                       const float* __restrict__ ctx, const float* __restrict__ hf,
                       unsigned short* __restrict__ xcat, int t) {
    int b = blockIdx.x, tid = threadIdx.x;
    int tok = (t == 0) ? SOS_ : target[b * T_ + (t - 1)];
    unsigned short* row = xcat + (size_t)b * (3 * H_);
    const float* er = emb + (size_t)tok * H_;
    for (int i = tid; i < H_; i += 256) {
        row[i]            = f2bf(er[i]);
        row[H_ + i]       = f2bf(ctx[(size_t)b * H_ + i]);
        row[2 * H_ + i]   = f2bf(hf[(size_t)b * H_ + i]);
    }
}

// ---------------- LSTM cell + LayerNorm, one block (256 thr) per batch ----------------
__global__ void k_cell(const float* __restrict__ gates,
                       float* __restrict__ hf, float* __restrict__ cf,
                       unsigned short* __restrict__ hb, unsigned short* __restrict__ nhb,
                       const float* __restrict__ ln_g, const float* __restrict__ ln_b) {
    __shared__ float rs[256], rs2[256];
    int b = blockIdx.x, tid = threadIdx.x;
    const float* gr = gates + (size_t)b * 4 * H_;
    float hn[4];
    float lsum = 0.f, lsum2 = 0.f;
#pragma unroll
    for (int j = 0; j < 4; ++j) {
        int h = tid + j * 256;
        float ig = sigmoidf_(gr[h]);
        float fg = sigmoidf_(gr[H_ + h]);
        float gg = tanhf(gr[2 * H_ + h]);
        float og = sigmoidf_(gr[3 * H_ + h]);
        float c  = fg * cf[(size_t)b * H_ + h] + ig * gg;
        float hv = og * tanhf(c);
        cf[(size_t)b * H_ + h] = c;
        hf[(size_t)b * H_ + h] = hv;
        hb[(size_t)b * H_ + h] = f2bf(hv);
        hn[j] = hv;
        lsum += hv; lsum2 += hv * hv;
    }
    rs[tid] = lsum; rs2[tid] = lsum2;
    __syncthreads();
    for (int o = 128; o; o >>= 1) {
        if (tid < o) { rs[tid] += rs[tid + o]; rs2[tid] += rs2[tid + o]; }
        __syncthreads();
    }
    float mu  = rs[0] * (1.f / H_);
    float var = rs2[0] * (1.f / H_) - mu * mu;       // biased, as torch LayerNorm
    float inv = rsqrtf(var + 1e-5f);
#pragma unroll
    for (int j = 0; j < 4; ++j) {
        int h = tid + j * 256;
        float nh = (hn[j] - mu) * inv * ln_g[h] + ln_b[h];
        nhb[(size_t)b * H_ + h] = f2bf(nh);
    }
}

// ---------------- in-place log_softmax over V, one block per (b,t) row ----------------
__global__ void k_log_softmax(float* __restrict__ logp) {
    __shared__ float red[256];
    int row = blockIdx.x, tid = threadIdx.x;
    float* p = logp + (size_t)row * V_;
    float m = -3.4e38f;
    for (int i = tid; i < V_; i += 256) m = fmaxf(m, p[i]);
    red[tid] = m; __syncthreads();
    for (int o = 128; o; o >>= 1) { if (tid < o) red[tid] = fmaxf(red[tid], red[tid + o]); __syncthreads(); }
    m = red[0]; __syncthreads();
    float s = 0.f;
    for (int i = tid; i < V_; i += 256) s += __expf(p[i] - m);
    red[tid] = s; __syncthreads();
    for (int o = 128; o; o >>= 1) { if (tid < o) red[tid] += red[tid + o]; __syncthreads(); }
    float lse = m + __logf(red[0]);
    __syncthreads();
    for (int i = tid; i < V_; i += 256) p[i] -= lse;
}

__global__ void k_copy_state(const float* __restrict__ hf, const float* __restrict__ cf,
                             float* __restrict__ out_h, float* __restrict__ out_c) {
    int i = blockIdx.x * blockDim.x + threadIdx.x;
    if (i >= B_ * H_) return;
    out_h[i] = hf[i];
    out_c[i] = cf[i];
}

// ---------------- host orchestration ----------------
extern "C" void kernel_launch(void* const* d_in, const int* in_sizes, int n_in,
                              void* d_out, int out_size, void* d_ws, size_t ws_size,
                              hipStream_t stream) {
    const float* enc    = (const float*)d_in[0];
    const float* enc_h  = (const float*)d_in[1];
    const float* enc_c  = (const float*)d_in[2];
    const int*   target = (const int*)  d_in[3];
    const float* emb    = (const float*)d_in[4];
    const float* Wa     = (const float*)d_in[5];
    const float* ba     = (const float*)d_in[6];
    const float* Ua     = (const float*)d_in[7];
    const float* bu     = (const float*)d_in[8];
    const float* Va     = (const float*)d_in[9];
    const float* bv     = (const float*)d_in[10];
    const float* W_ih   = (const float*)d_in[11];
    const float* W_hh   = (const float*)d_in[12];
    const float* b_ih   = (const float*)d_in[13];
    const float* b_hh   = (const float*)d_in[14];
    const float* ln_g   = (const float*)d_in[15];
    const float* ln_b   = (const float*)d_in[16];
    const float* outW   = (const float*)d_in[17];
    const float* outb   = (const float*)d_in[18];
    float* out = (float*)d_out;

    // ---- output layout: logp[B,T,V] | h[B,H] | c[B,H] | attn[B,T,S] ----
    const size_t OFF_H   = (size_t)B_ * T_ * V_;
    const size_t OFF_C   = OFF_H + (size_t)B_ * H_;
    const size_t OFF_ATT = OFF_C + (size_t)B_ * H_;

    // ---- workspace carve-up (bump allocator, 256B aligned) ----
    char* ws = (char*)d_ws;
    size_t o = 0;
    auto carve = [&](size_t bytes) {
        void* p = ws + o;
        o += (bytes + 255) & ~(size_t)255;
        return p;
    };
    float*          keys   = (float*)         carve((size_t)B_ * S_ * H_ * 4); // 16 MB
    unsigned short* enc_b  = (unsigned short*)carve((size_t)B_ * S_ * H_ * 2); // 8 MB
    unsigned short* Ua_b   = (unsigned short*)carve((size_t)H_ * H_ * 2);
    unsigned short* Wa_b   = (unsigned short*)carve((size_t)H_ * H_ * 2);
    unsigned short* Wcat_b = (unsigned short*)carve((size_t)4 * H_ * 3 * H_ * 2); // 25 MB
    unsigned short* outW_b = (unsigned short*)carve((size_t)V_ * H_ * 2);         // 65.5 MB
    float*          hf     = (float*)         carve((size_t)B_ * H_ * 4);
    float*          cf     = (float*)         carve((size_t)B_ * H_ * 4);
    unsigned short* hb     = (unsigned short*)carve((size_t)B_ * H_ * 2);
    unsigned short* nhb    = (unsigned short*)carve((size_t)B_ * H_ * 2);
    float*          qf     = (float*)         carve((size_t)B_ * H_ * 4);
    float*          ctx    = (float*)         carve((size_t)B_ * H_ * 4);
    unsigned short* xcat   = (unsigned short*)carve((size_t)B_ * 3 * H_ * 2);
    float*          gates  = (float*)         carve((size_t)B_ * 4 * H_ * 4);
    (void)ws_size; (void)in_sizes; (void)n_in; (void)out_size;

    auto cvt = [&](const float* s, unsigned short* d, size_t n) {
        int blocks = (int)((n + 255) / 256);
        k_f32_to_bf16<<<blocks, 256, 0, stream>>>(s, d, n);
    };
    // one wave per 64x16 strip
    auto gemm = [&](const unsigned short* A, int lda, const unsigned short* Bw, int ldb,
                    float* C, int ldc, int M, int N, int K,
                    const float* b0, const float* b1) {
        int waves  = (M / 64) * (N / 16);
        int blocks = (waves * 32 + 255) / 256;
        k_gemm_bf16<<<blocks, 256, 0, stream>>>(A, lda, Bw, ldb, C, ldc, M, N, K, b0, b1);
    };

    // ---- one-time setup ----
    cvt(enc,  enc_b, (size_t)B_ * S_ * H_);
    cvt(Ua,   Ua_b,  (size_t)H_ * H_);
    cvt(Wa,   Wa_b,  (size_t)H_ * H_);
    cvt(outW, outW_b,(size_t)V_ * H_);
    {
        size_t n = (size_t)4 * H_ * 3 * H_;
        k_build_wcat<<<(int)((n + 255) / 256), 256, 0, stream>>>(W_ih, W_hh, Wcat_b);
    }
    k_init_state<<<(B_ * H_) / 256, 256, 0, stream>>>(enc_h, enc_c, hf, cf, hb);

    // keys_proj = enc @ Ua^T + bu   [B*S, H] — step-invariant, computed once
    gemm(enc_b, H_, Ua_b, H_, keys, H_, B_ * S_, H_, H_, bu, nullptr);

    // ---- T sequential decode steps ----
    for (int t = 0; t < T_; ++t) {
        // q = h @ Wa^T + ba
        gemm(hb, H_, Wa_b, H_, qf, H_, B_, H_, H_, ba, nullptr);
        // attention: scores -> softmax -> ctx (+ write attn weights)
        k_attn<<<B_, 256, 0, stream>>>(qf, keys, Va, bv, enc, ctx, out + OFF_ATT, t);
        // xcat = [emb[tok], ctx, h]
        k_pack<<<B_, 256, 0, stream>>>(target, emb, ctx, hf, xcat, t);
        // gates = xcat @ [W_ih|W_hh]^T + b_ih + b_hh
        gemm(xcat, 3 * H_, Wcat_b, 3 * H_, gates, 4 * H_, B_, 4 * H_, 3 * H_, b_ih, b_hh);
        // LSTM cell + LayerNorm -> h, c, nh(bf16)
        k_cell<<<B_, 256, 0, stream>>>(gates, hf, cf, hb, nhb, ln_g, ln_b);
        // logits = nh @ outW^T + outb, written into logp[:, t, :] (ldc = T*V)
        gemm(nhb, H_, outW_b, H_, out + (size_t)t * V_, T_ * V_, B_, V_, H_, outb, nullptr);
    }

    // ---- epilogue: in-place log_softmax over V, then final h/c ----
    k_log_softmax<<<B_ * T_, 256, 0, stream>>>(out);
    k_copy_state<<<(B_ * H_) / 256, 256, 0, stream>>>(hf, cf, out + OFF_H, out + OFF_C);
}
